// Block_4526895530469
// MI455X (gfx1250) — compile-verified, matrix operand order checked by reading, not run on hardware
//
#include <hip/hip_runtime.h>

// ---------------------------------------------------------------------------
// Mamba block + MLP for MI455X (gfx1250), wave32, WMMA bf16 GEMM path.
// D_MODEL=1024, D_INNER=2048, D_STATE=16, D_CONV=4, DT_RANK=64, B=2, T=2048
// ---------------------------------------------------------------------------

typedef __bf16 bf16;
typedef __attribute__((ext_vector_type(16))) __bf16 v16bf;
typedef __attribute__((ext_vector_type(8)))  __bf16 v8bf;
typedef __attribute__((ext_vector_type(8)))  float  v8f;

#define NT    4096   // B*T token rows
#define TSEQ  2048
#define DM    1024
#define DE    2048
#define DS    16
#define DR    64
#define DF    96     // DT_RANK + 2*D_STATE

// ---------------------------------------------------------------------------
// bf16 WMMA fragment load: operand row-major, K contiguous (ld = K).
// Lane r=lane&15 selects the row, lane>>4 selects the K-octet; fragment
// elements 0..7 = K base..base+7, elements 8..15 = K 16+base..16+base+7.
// ---------------------------------------------------------------------------
__device__ __forceinline__ v16bf load_frag(const bf16* __restrict__ p) {
  v8bf lo = *reinterpret_cast<const v8bf*>(p);
  v8bf hi = *reinterpret_cast<const v8bf*>(p + 16);
  return __builtin_shufflevector(lo, hi, 0, 1, 2, 3, 4, 5, 6, 7,
                                         8, 9, 10, 11, 12, 13, 14, 15);
}

// C[M,N](f32) = A[M,K](bf16,row-major) * B[N,K]^T(bf16,row-major).
// 256 threads = 8 waves.  Wave tile = (16*MT) x (16*JT); block tile
// = (128*MT) x (16*JT).  Fully unrolled, branch-free inner loop.
template <int MT, int JT>
__global__ __launch_bounds__(256) void k_gemm_bf16(
    float* __restrict__ C, const bf16* __restrict__ A, const bf16* __restrict__ B,
    int M, int N, int K)
{
  const int lane = threadIdx.x & 31;
  const int wave = threadIdx.x >> 5;
  const int m0   = blockIdx.y * (128 * MT) + wave * (16 * MT);
  const int nb   = blockIdx.x * (16 * JT);
  const int r    = lane & 15;
  const int half = lane >> 4;

  const bf16* Arow[MT];
#pragma unroll
  for (int mi = 0; mi < MT; ++mi)
    Arow[mi] = A + (size_t)(m0 + mi * 16 + r) * K + half * 8;
  const bf16* Brow[JT];
#pragma unroll
  for (int j = 0; j < JT; ++j)
    Brow[j] = B + (size_t)(nb + j * 16 + r) * K + half * 8;

  v8f acc[MT][JT];
#pragma unroll
  for (int mi = 0; mi < MT; ++mi)
#pragma unroll
    for (int j = 0; j < JT; ++j) acc[mi][j] = v8f{};

  for (int k0 = 0; k0 < K; k0 += 32) {
    v16bf a[MT];
#pragma unroll
    for (int mi = 0; mi < MT; ++mi) a[mi] = load_frag(Arow[mi] + k0);
    v16bf b[JT];
#pragma unroll
    for (int j = 0; j < JT; ++j) b[j] = load_frag(Brow[j] + k0);
#pragma unroll
    for (int mi = 0; mi < MT; ++mi)
#pragma unroll
      for (int j = 0; j < JT; ++j)
        acc[mi][j] = __builtin_amdgcn_wmma_f32_16x16x32_bf16(
            false, a[mi], false, b[j], (short)0, acc[mi][j], false, false);
  }

  // D layout: VGPR i holds rows m0+i (lanes 0-15) and m0+8+i (lanes 16-31),
  // column = n0 + (lane&15).
#pragma unroll
  for (int mi = 0; mi < MT; ++mi)
#pragma unroll
    for (int j = 0; j < JT; ++j) {
      float* crow = C + (size_t)(m0 + mi * 16 + half * 8) * N + nb + j * 16 + r;
#pragma unroll
      for (int i = 0; i < 8; ++i) crow[(size_t)i * N] = acc[mi][j][i];
    }
}

// ---------------------------------------------------------------------------
// rmsnorm over last dim (ncols) -> bf16 output.  One block per row.
// ---------------------------------------------------------------------------
__global__ __launch_bounds__(256) void k_rmsnorm_bf16(
    bf16* __restrict__ o, const float* __restrict__ x, int ncols)
{
  __shared__ float red[256];
  const int row = blockIdx.x;
  const float* xr = x + (size_t)row * ncols;
  float ss = 0.f;
  for (int c = threadIdx.x; c < ncols; c += 256) { float v = xr[c]; ss += v * v; }
  red[threadIdx.x] = ss;
  __syncthreads();
  for (int off = 128; off > 0; off >>= 1) {
    if (threadIdx.x < off) red[threadIdx.x] += red[threadIdx.x + off];
    __syncthreads();
  }
  const float rs = rsqrtf(red[0] / (float)ncols + 1.1920928955078125e-07f);
  bf16* orow = o + (size_t)row * ncols;
  for (int c = threadIdx.x; c < ncols; c += 256) orow[c] = (bf16)(xr[c] * rs);
}

// f32 -> bf16 convert (weights, each call; deterministic)
__global__ __launch_bounds__(256) void k_cvt_bf16(
    bf16* __restrict__ o, const float* __restrict__ in, int n)
{
  int i = blockIdx.x * 256 + threadIdx.x;
  if (i < n) o[i] = (bf16)in[i];
}

// Depthwise causal conv (D_CONV=4) over the x-half of xz, + bias, + silu.
// Writes f32 (for scan) and bf16 (for x_proj GEMM).
__global__ __launch_bounds__(256) void k_conv_silu(
    float* __restrict__ xcf, bf16* __restrict__ xcb,
    const float* __restrict__ xz, const float* __restrict__ cw,
    const float* __restrict__ cb)
{
  const int n = blockIdx.x;            // token row (b*T + t)
  const int t = n & (TSEQ - 1);
  const int e = blockIdx.y * 256 + threadIdx.x;
  const size_t base = (size_t)n * (2 * DE) + e;   // xz row stride = 4096
  float acc = cb[e];
  const float w0 = cw[e * 4 + 0], w1 = cw[e * 4 + 1],
              w2 = cw[e * 4 + 2], w3 = cw[e * 4 + 3];
  if (t >= 3) acc += w0 * xz[base - 3 * (size_t)(2 * DE)];
  if (t >= 2) acc += w1 * xz[base - 2 * (size_t)(2 * DE)];
  if (t >= 1) acc += w2 * xz[base - 1 * (size_t)(2 * DE)];
  acc += w3 * xz[base];
  const float sv = acc / (1.f + __expf(-acc));    // silu
  xcf[(size_t)n * DE + e] = sv;
  xcb[(size_t)n * DE + e] = (bf16)sv;
}

// Extract dt_low (first 64 cols of x_dbl) as bf16 for the dt_proj GEMM.
__global__ __launch_bounds__(256) void k_dtlow(
    bf16* __restrict__ o, const float* __restrict__ xdbl, int n)
{
  int i = blockIdx.x * 256 + threadIdx.x;
  if (i < n) {
    int row = i >> 6, c = i & 63;
    o[i] = (bf16)xdbl[(size_t)row * DF + c];
  }
}

// dt = softplus(dt_pre + bias), in place.
__global__ __launch_bounds__(256) void k_softplus(
    float* __restrict__ d, const float* __restrict__ bias, int n)
{
  int i = blockIdx.x * 256 + threadIdx.x;
  if (i < n) {
    float v = d[i] + bias[i & (DE - 1)];
    d[i] = (v > 20.f) ? v : log1pf(__expf(v));
  }
}

// ---------------------------------------------------------------------------
// Selective scan: one thread owns one (b, e) channel, 16 states in registers,
// sequential over t.  B_t / C_t (32 floats, shared by all channels of batch b)
// staged via LDS once per step.  Output ys = y_t + u*D (pre-gate).
// ---------------------------------------------------------------------------
__global__ __launch_bounds__(256) void k_scan(
    float* __restrict__ ys, const float* __restrict__ dt,
    const float* __restrict__ xdbl, const float* __restrict__ xconv,
    const float* __restrict__ A_log, const float* __restrict__ Dv)
{
  __shared__ float sBC[32];
  const int b = blockIdx.y;
  const int e = blockIdx.x * 256 + threadIdx.x;
  float An[DS], h[DS];
#pragma unroll
  for (int s = 0; s < DS; ++s) {
    An[s] = -__expf(A_log[(size_t)e * DS + s]);   // A = -exp(A_log)
    h[s]  = 0.f;
  }
  const float Dn = Dv[e];

  for (int t = 0; t < TSEQ; ++t) {
    const size_t n = (size_t)b * TSEQ + t;
    if (threadIdx.x < 32) sBC[threadIdx.x] = xdbl[n * DF + DR + threadIdx.x];
    __syncthreads();
    const float dtv = dt[n * DE + e];
    const float u   = xconv[n * DE + e];
    const float du  = dtv * u;
    float y = 0.f;
#pragma unroll
    for (int s = 0; s < DS; ++s) {
      h[s] = h[s] * __expf(dtv * An[s]) + du * sBC[s];   // h*exp(dt*A)+dt*u*B
      y   += h[s] * sBC[DS + s];                          // sum_s h*C
    }
    ys[n * DE + e] = y + u * Dn;
    __syncthreads();
  }
}

// y_gated(bf16) = ys * silu(z), z = xz[:, 2048+e]
__global__ __launch_bounds__(256) void k_gate(
    bf16* __restrict__ yb, const float* __restrict__ ys,
    const float* __restrict__ xz, int n)
{
  int i = blockIdx.x * 256 + threadIdx.x;
  if (i < n) {
    int row = i >> 11, e = i & (DE - 1);
    float z = xz[(size_t)row * (2 * DE) + DE + e];
    float g = z / (1.f + __expf(-z));
    yb[i] = (bf16)(ys[i] * g);
  }
}

// h2(bf16) = relu(v)^2
__global__ __launch_bounds__(256) void k_relusq(
    bf16* __restrict__ o, const float* __restrict__ in, int n)
{
  int i = blockIdx.x * 256 + threadIdx.x;
  if (i < n) { float v = fmaxf(in[i], 0.f); o[i] = (bf16)(v * v); }
}

__global__ __launch_bounds__(256) void k_add(
    float* __restrict__ o, const float* __restrict__ a,
    const float* __restrict__ b, int n)
{
  int i = blockIdx.x * 256 + threadIdx.x;
  if (i < n) o[i] = a[i] + b[i];
}

// ---------------------------------------------------------------------------
extern "C" void kernel_launch(void* const* d_in, const int* in_sizes, int n_in,
                              void* d_out, int out_size, void* d_ws, size_t ws_size,
                              hipStream_t stream) {
  const float* x         = (const float*)d_in[0];
  const float* in_proj_w = (const float*)d_in[1];
  const float* conv_w    = (const float*)d_in[2];
  const float* conv_b    = (const float*)d_in[3];
  const float* x_proj_w  = (const float*)d_in[4];
  const float* dt_proj_w = (const float*)d_in[5];
  const float* dt_proj_b = (const float*)d_in[6];
  const float* A_log     = (const float*)d_in[7];
  const float* Dv        = (const float*)d_in[8];
  const float* out_proj_w= (const float*)d_in[9];
  const float* c_fc_w    = (const float*)d_in[10];
  const float* c_proj_w  = (const float*)d_in[11];
  float* out = (float*)d_out;

  char* ws = (char*)d_ws;
  const size_t MB = 1u << 20;
  bf16*  xn_bf  = (bf16*) (ws +   0 * MB);  //  8MB  (aliased: h_bf for MLP)
  bf16*  wbuf   = (bf16*) (ws +   8 * MB);  //  8MB  shared bf16 weight scratch
  float* xz     = (float*)(ws +  16 * MB);  // 64MB  in_proj output (x|z)
  float* xconv  = (float*)(ws +  80 * MB);  // 32MB  conv+silu f32
  bf16*  xconvb = (bf16*) (ws + 112 * MB);  // 16MB  (aliased: y_gated bf16)
  float* xdbl   = (float*)(ws + 128 * MB);  //  2MB  x_proj output (96 cols)
  bf16*  dtlowb = (bf16*) (ws + 130 * MB);  //  1MB
  float* dtbuf  = (float*)(ws + 131 * MB);  // 32MB  dt (aliased: c_fc out)
  float* ysbuf  = (float*)(ws + 163 * MB);  // 32MB  scan out (aliased: relu^2 bf16)
  float* tmpd   = (float*)(ws + 195 * MB);  // 16MB  out_proj / c_proj f32 out
  float* xres   = (float*)(ws + 211 * MB);  // 16MB  residual stream
  bf16*  yb     = xconvb;
  bf16*  h2b    = (bf16*)ysbuf;

  const int EW = NT * DE;   // 8.4M elements

  // ---- mamba branch ------------------------------------------------------
  k_rmsnorm_bf16<<<NT, 256, 0, stream>>>(xn_bf, x, DM);

  k_cvt_bf16<<<(2 * DE * DM + 255) / 256, 256, 0, stream>>>(wbuf, in_proj_w, 2 * DE * DM);
  // 4096 x 4096 x 1024: block tile 256x64
  k_gemm_bf16<2, 4><<<dim3(64, 16), 256, 0, stream>>>(xz, xn_bf, wbuf, NT, 2 * DE, DM);

  k_conv_silu<<<dim3(NT, DE / 256), 256, 0, stream>>>(xconv, xconvb, xz, conv_w, conv_b);

  k_cvt_bf16<<<(DF * DE + 255) / 256, 256, 0, stream>>>(wbuf, x_proj_w, DF * DE);
  // 4096 x 96 x 2048: wave tile 16x32, block tile 128x32, grid 3x32
  k_gemm_bf16<1, 2><<<dim3(3, 32), 256, 0, stream>>>(xdbl, xconvb, wbuf, NT, DF, DE);

  k_dtlow<<<(NT * DR + 255) / 256, 256, 0, stream>>>(dtlowb, xdbl, NT * DR);
  k_cvt_bf16<<<(DE * DR + 255) / 256, 256, 0, stream>>>(wbuf, dt_proj_w, DE * DR);
  // 4096 x 2048 x 64
  k_gemm_bf16<2, 4><<<dim3(32, 16), 256, 0, stream>>>(dtbuf, dtlowb, wbuf, NT, DE, DR);
  k_softplus<<<(EW + 255) / 256, 256, 0, stream>>>(dtbuf, dt_proj_b, EW);

  k_scan<<<dim3(DE / 256, 2), 256, 0, stream>>>(ysbuf, dtbuf, xdbl, xconv, A_log, Dv);
  k_gate<<<(EW + 255) / 256, 256, 0, stream>>>(yb, ysbuf, xz, EW);

  k_cvt_bf16<<<(DM * DE + 255) / 256, 256, 0, stream>>>(wbuf, out_proj_w, DM * DE);
  // 4096 x 1024 x 2048
  k_gemm_bf16<2, 4><<<dim3(16, 16), 256, 0, stream>>>(tmpd, yb, wbuf, NT, DM, DE);
  k_add<<<(NT * DM + 255) / 256, 256, 0, stream>>>(xres, x, tmpd, NT * DM);

  // ---- MLP branch --------------------------------------------------------
  k_rmsnorm_bf16<<<NT, 256, 0, stream>>>(xn_bf, xres, DM);

  k_cvt_bf16<<<(2 * DM * DM + 255) / 256, 256, 0, stream>>>(wbuf, c_fc_w, 2 * DM * DM);
  // 4096 x 2048 x 1024
  k_gemm_bf16<2, 4><<<dim3(32, 16), 256, 0, stream>>>(dtbuf, xn_bf, wbuf, NT, 2 * DM, DM);
  k_relusq<<<(EW + 255) / 256, 256, 0, stream>>>(h2b, dtbuf, EW);

  k_cvt_bf16<<<(DM * 2 * DM + 255) / 256, 256, 0, stream>>>(wbuf, c_proj_w, DM * 2 * DM);
  // 4096 x 1024 x 2048
  k_gemm_bf16<2, 4><<<dim3(16, 16), 256, 0, stream>>>(tmpd, h2b, wbuf, NT, DM, 2 * DM);
  k_add<<<(NT * DM + 255) / 256, 256, 0, stream>>>(out, xres, tmpd, NT * DM);

  (void)in_sizes; (void)n_in; (void)out_size; (void)ws_size;
}
